// vi_image_model_22316650070876
// MI455X (gfx1250) — matrix operand order repaired and checked by reading, not compiled, not run on hardware
//
// CDNA5 / gfx1250 implementation of a 100-sample Bayesian CNN MC forward pass.
// All matrix math (conv1/conv2/fc1/fc2, ~99.8% of 59 GFLOP) runs on
// v_wmma_f32_16x16x32_bf16 with f32 accumulation. Weights are materialized in
// zero-padded bf16 layouts so every GEMM tile feed is an unconditional aligned
// 16B copy -> GLOBAL_LOAD_ASYNC_TO_LDS_B128 (ASYNCcnt) where the toolchain
// exposes it; im2col patches are built branchlessly in LDS so A/B fragments are
// pure ds_load_b128. HBM traffic ~350MB (epsilon-read dominated, ~15us @ 23.3TB/s).
#include <hip/hip_runtime.h>
#include <hip/hip_bf16.h>
#include <math.h>

#define S_  100
#define B_  128
#define NW_ 430500
#define NB_ 580

#if __has_builtin(__builtin_amdgcn_global_load_async_to_lds_b128) && \
    __has_builtin(__builtin_amdgcn_s_wait_asynccnt)
#define USE_ASYNC 1
#else
#define USE_ASYNC 0
#endif

typedef __attribute__((ext_vector_type(16))) __bf16 v16bf;
typedef __attribute__((ext_vector_type(8)))  float  v8f;
typedef __attribute__((ext_vector_type(8)))  unsigned short v8u16;
typedef int v4i_vs __attribute__((vector_size(4 * sizeof(int))));
typedef __attribute__((address_space(1))) v4i_vs* gvec_p;   // global 16B vector ptr
typedef __attribute__((address_space(3))) v4i_vs* lvec_p;   // LDS 16B vector ptr

__device__ __forceinline__ unsigned short f2bf(float f) {
  unsigned u = __float_as_uint(f);
  u += 0x7fffu + ((u >> 16) & 1u);           // round-to-nearest-even
  return (unsigned short)(u >> 16);
}
__device__ __forceinline__ __bf16 bfbits(unsigned short u) {
  return __builtin_bit_cast(__bf16, u);
}
__device__ __forceinline__ v8f zero8() {
  v8f z;
#pragma unroll
  for (int i = 0; i < 8; ++i) z[i] = 0.0f;
  return z;
}
__device__ __forceinline__ v8f wmma_bf16(v16bf a, v16bf b, v8f c) {
  return __builtin_amdgcn_wmma_f32_16x16x32_bf16(false, a, false, b, (short)0, c,
                                                 false, false);
}
// 16B global->LDS copy: async-to-LDS when available, else plain vector copy.
__device__ __forceinline__ void cp16(const unsigned short* g, unsigned short* l) {
#if USE_ASYNC
  unsigned short* gm = const_cast<unsigned short*>(g);
  __builtin_amdgcn_global_load_async_to_lds_b128((gvec_p)gm, (lvec_p)l, 0, 0);
#else
  *(v8u16*)l = *(const v8u16*)g;
#endif
}
template <int N> __device__ __forceinline__ void wait_async() {
#if USE_ASYNC
  __builtin_amdgcn_s_wait_asynccnt(N);
#endif
}
// A fragment (16x32 bf16, contiguous K = 16*half + s) from row-major LDS tile, stride ld.
__device__ __forceinline__ v16bf load_a_frag(const unsigned short* t, int ld, int m0) {
  const int lane = threadIdx.x & 31, lh = lane >> 4;
  const unsigned short* r = t + (m0 + (lane & 15)) * ld;
  v16bf a;
#pragma unroll
  for (int i = 0; i < 8; ++i) {
    a[i]     = bfbits(r[16 * lh + i]);       // K = 16*half + 0..7
    a[8 + i] = bfbits(r[16 * lh + 8 + i]);   // K = 16*half + 8..15
  }
  return a;
}
// B fragment (32x16 bf16) from an LDS B^T tile (rows = N, K contiguous).
__device__ __forceinline__ v16bf load_bt_frag(const unsigned short* t, int ld, int n0) {
  const int lane = threadIdx.x & 31, lh = lane >> 4;
  const unsigned short* r = t + (n0 + (lane & 15)) * ld + 16 * lh; // K = 16*half+0..15
  v16bf b;
#pragma unroll
  for (int i = 0; i < 16; ++i) b[i] = bfbits(r[i]);
  return b;
}
__device__ __forceinline__ float softplusf(float r) {
  return (r > 20.0f) ? r : log1pf(__expf(r));
}

// ---------------- Stage 0: softplus(rho) once -------------------------------
__global__ __launch_bounds__(256) void k_softplus(const float* __restrict__ rho_w,
                                                  const float* __restrict__ rho_b,
                                                  float* __restrict__ sp_w,
                                                  float* __restrict__ sp_b) {
  const int i = blockIdx.x * 256 + threadIdx.x;
  if (i < NW_) sp_w[i] = softplusf(rho_w[i]);
  if (i < NB_) sp_b[i] = softplusf(rho_b[i]);
}

// ---------------- Stage 1: per-sample weights -> padded bf16 layouts --------
// w1p [s][32][32], w2p [s][64][512], w3p [s][512][800], w4p [s][16][512];
// pads pre-zeroed by hipMemsetAsync, so GEMM tile copies are unconditional.
__global__ __launch_bounds__(256) void k_transform(const float* __restrict__ e,
                                                   const float* __restrict__ mu_w,
                                                   const float* __restrict__ sp_w,
                                                   const float* __restrict__ mu_b,
                                                   const float* __restrict__ sp_b,
                                                   unsigned short* __restrict__ w1p,
                                                   unsigned short* __restrict__ w2p,
                                                   unsigned short* __restrict__ w3p,
                                                   unsigned short* __restrict__ w4p,
                                                   float* __restrict__ bia) {
  const int s = blockIdx.y;
  const float* es = e + (size_t)s * (NW_ + NB_);
  for (int i = blockIdx.x * 256 + threadIdx.x; i < NW_; i += gridDim.x * 256) {
    const unsigned short v = f2bf(fmaf(sp_w[i], es[i], mu_w[i]));
    if (i < 500) {                                   // conv1 [20][25]
      w1p[(size_t)s * 1024 + (i / 25) * 32 + i % 25] = v;
    } else if (i < 25500) {                          // conv2 [50][500]
      const int j = i - 500;
      w2p[(size_t)s * 32768 + (j / 500) * 512 + j % 500] = v;
    } else if (i < 425500) {                         // fc1 [500][800] (cols exact)
      w3p[(size_t)s * 409600 + (i - 25500)] = v;
    } else {                                         // fc2 [10][500]
      const int j = i - 425500;
      w4p[(size_t)s * 8192 + (j / 500) * 512 + j % 500] = v;
    }
  }
  const int j = blockIdx.x * 256 + threadIdx.x;
  if (j < NB_) bia[s * NB_ + j] = fmaf(sp_b[j], es[NW_ + j], mu_b[j]);
}

// ---------------- Stage 2: conv1 (1->20, 5x5) + relu + 2x2 maxpool ----------
// Implicit GEMM per (b, s, half): M=288 spatial, N=32 (20 pad), K=32 (25 pad).
__global__ __launch_bounds__(128) void k_conv1(const float* __restrict__ x,
                                               const unsigned short* __restrict__ w1p,
                                               const float* __restrict__ bia,
                                               unsigned short* __restrict__ h1) {
  const int b = blockIdx.x, s = blockIdx.y, hh = blockIdx.z;
  __shared__ alignas(16) unsigned short simg[28 * 32];
  __shared__ alignas(16) unsigned short swt[32 * 32];
  __shared__ alignas(16) unsigned short spat[288 * 32];
  __shared__ alignas(16) float sout[288 * 32];
  const int t = threadIdx.x;
  cp16(w1p + (size_t)s * 1024 + t * 8, swt + t * 8);      // 128 x 16B
  const float* xb = x + b * 784;
  for (int i = t; i < 784; i += 128) simg[(i / 28) * 32 + (i % 28)] = f2bf(xb[i]);
  wait_async<0>();
  __syncthreads();
  {  // branchless im2col: each thread owns patch column j (constant ky/kx/valid)
    const int j = t & 31, m0 = t >> 5;
    const bool valid = j < 25;
    const int ky = valid ? j / 5 : 0, kx = valid ? j % 5 : 0;
    for (int m = m0; m < 288; m += 4) {
      const unsigned y = (unsigned)m / 24u, xx = (unsigned)m % 24u;
      const unsigned short v = simg[(hh * 12 + y + ky) * 32 + xx + kx];
      spat[m * 32 + j] = valid ? v : (unsigned short)0;
    }
  }
  __syncthreads();
  const int wv = t >> 5, lane = t & 31, lh = lane >> 4, lr = lane & 15;
  for (int mt = wv; mt < 18; mt += 4) {      // wave-uniform (EXEC all ones at WMMA)
    const v16bf a = load_a_frag(spat, 32, mt * 16);
#pragma unroll
    for (int nt = 0; nt < 2; ++nt) {
      v8f c = zero8();
      c = wmma_bf16(a, load_bt_frag(swt, 32, nt * 16), c);
#pragma unroll
      for (int r = 0; r < 8; ++r)
        sout[(mt * 16 + 8 * lh + r) * 32 + nt * 16 + lr] = c[r];
    }
  }
  __syncthreads();
  for (int i = t; i < 1440; i += 128) {      // 20co x 6y2 x 12x2 ; relu(max)=max(relu)
    const int co = i / 72, rem = i % 72, y2l = rem / 12, x2 = rem % 12;
    const int m00 = (2 * y2l) * 24 + 2 * x2;
    float mx = fmaxf(fmaxf(sout[m00 * 32 + co], sout[(m00 + 1) * 32 + co]),
                     fmaxf(sout[(m00 + 24) * 32 + co], sout[(m00 + 25) * 32 + co]));
    mx = fmaxf(mx + bia[s * NB_ + co], 0.0f);
    const int y2 = hh * 6 + y2l;
    h1[((size_t)(s * B_ + b) * 20 + co) * 144 + y2 * 12 + x2] = f2bf(mx);
  }
}

// ---------------- Stage 3: conv2 (20->50, 5x5) + relu + 2x2 maxpool ---------
// Per (b,s): M=64 (8x8), N=64 (50 pad), K=512 (500 pad), K streamed in 128-chunks.
__global__ __launch_bounds__(128) void k_conv2(const unsigned short* __restrict__ h1,
                                               const unsigned short* __restrict__ w2p,
                                               const float* __restrict__ bia,
                                               unsigned short* __restrict__ h2) {
  const int b = blockIdx.x, s = blockIdx.y;
  __shared__ alignas(16) unsigned short simg[2880];       // 20x12x12 input
  __shared__ alignas(16) unsigned short sbt[64 * 128];    // W2^T K-chunk
  __shared__ alignas(16) unsigned short spat[64 * 128];   // im2col K-chunk
  __shared__ alignas(16) float sout[64 * 64];
  const int t = threadIdx.x;
  const unsigned short* src = h1 + (size_t)(s * B_ + b) * 2880;
  for (int v = t; v < 360; v += 128) cp16(src + v * 8, simg + v * 8);
  const unsigned short* w2s = w2p + (size_t)s * 32768;
  const int wv = t >> 5, lane = t & 31, lh = lane >> 4, lr = lane & 15;
  const int moff = (wv >> 1) * 32, noff = (wv & 1) * 32;
  v8f acc[4];
#pragma unroll
  for (int q = 0; q < 4; ++q) acc[q] = zero8();
  for (int c = 0; c < 4; ++c) {
    __syncthreads();                          // prev chunk's readers done
    for (int v = t; v < 1024; v += 128) {     // 64x128 weight chunk, 16B vectors
      const int row = v >> 3, c8 = (v & 7) * 8;
      cp16(w2s + row * 512 + c * 128 + c8, sbt + row * 128 + c8);
    }
    wait_async<0>();
    __syncthreads();                          // simg + sbt resident
    {  // branchless im2col chunk: thread owns column t -> constant (ci,ky,kx)
      const int k = c * 128 + t;
      const bool valid = k < 500;
      const int kc = valid ? k : 0;
      const int ci = kc / 25, rr = kc % 25;
      const unsigned short* ib = simg + ci * 144 + (rr / 5) * 12 + (rr % 5);
      for (int m = 0; m < 64; ++m) {
        const unsigned short v = ib[(m >> 3) * 12 + (m & 7)];
        spat[m * 128 + t] = valid ? v : (unsigned short)0;
      }
    }
    __syncthreads();
    for (int sub = 0; sub < 4; ++sub) {
      const v16bf a0 = load_a_frag(spat + sub * 32, 128, moff);
      const v16bf a1 = load_a_frag(spat + sub * 32, 128, moff + 16);
      const v16bf b0 = load_bt_frag(sbt + sub * 32, 128, noff);
      const v16bf b1 = load_bt_frag(sbt + sub * 32, 128, noff + 16);
      acc[0] = wmma_bf16(a0, b0, acc[0]);
      acc[1] = wmma_bf16(a0, b1, acc[1]);
      acc[2] = wmma_bf16(a1, b0, acc[2]);
      acc[3] = wmma_bf16(a1, b1, acc[3]);
    }
  }
#pragma unroll
  for (int am = 0; am < 2; ++am)
#pragma unroll
    for (int bn = 0; bn < 2; ++bn)
#pragma unroll
      for (int r = 0; r < 8; ++r)
        sout[(moff + am * 16 + 8 * lh + r) * 64 + noff + bn * 16 + lr] = acc[am * 2 + bn][r];
  __syncthreads();
  for (int i = t; i < 800; i += 128) {        // i == co*16 + y2*4 + x2 (ref flatten)
    const int co = i >> 4, p = i & 15, y2 = p >> 2, x2 = p & 3;
    const int m00 = (2 * y2) * 8 + 2 * x2;
    float mx = fmaxf(fmaxf(sout[m00 * 64 + co], sout[(m00 + 1) * 64 + co]),
                     fmaxf(sout[(m00 + 8) * 64 + co], sout[(m00 + 9) * 64 + co]));
    mx = fmaxf(mx + bia[s * NB_ + 20 + co], 0.0f);
    h2[(size_t)(s * B_ + b) * 800 + i] = f2bf(mx);
  }
}

// ---------------- Stage 4: FC1 800->500 + relu, double-buffered async -------
__device__ __forceinline__ void aff1_issue(const unsigned short* h2s,
                                           const unsigned short* w3s, int kb,
                                           unsigned short* sa, unsigned short* sb, int t) {
#pragma unroll
  for (int q = 0; q < 2; ++q) {               // 4 async instrs per wave total
    const int v = t + q * 128, row = v >> 2, c8 = (v & 3) * 8;
    cp16(h2s + (size_t)row * 800 + kb + c8, sa + row * 32 + c8);
    cp16(w3s + (size_t)row * 800 + kb + c8, sb + row * 32 + c8);
  }
}
__global__ __launch_bounds__(128) void k_aff1(const unsigned short* __restrict__ h2,
                                              const unsigned short* __restrict__ w3p,
                                              const float* __restrict__ bia,
                                              unsigned short* __restrict__ h3) {
  const int mblk = blockIdx.x, nblk = blockIdx.y, s = blockIdx.z;
  __shared__ alignas(16) unsigned short sa[2][64 * 32];
  __shared__ alignas(16) unsigned short sb[2][64 * 32];
  const int t = threadIdx.x;
  const unsigned short* h2s = h2 + (size_t)(s * B_ + mblk * 64) * 800;
  const unsigned short* w3s = w3p + (size_t)s * 409600 + (size_t)nblk * 64 * 800;
  const int wv = t >> 5, lane = t & 31, lh = lane >> 4, lr = lane & 15;
  const int moff = (wv >> 1) * 32, noff = (wv & 1) * 32;
  v8f acc[4];
#pragma unroll
  for (int q = 0; q < 4; ++q) acc[q] = zero8();
  aff1_issue(h2s, w3s, 0, sa[0], sb[0], t);
  for (int kb = 0; kb < 800; kb += 32) {
    const int cur = (kb >> 5) & 1;
    if (kb + 32 < 800) {                      // overlap: issue next, wait current only
      aff1_issue(h2s, w3s, kb + 32, sa[cur ^ 1], sb[cur ^ 1], t);
      wait_async<4>();
    } else {
      wait_async<0>();
    }
    __syncthreads();
    const v16bf a0 = load_a_frag(sa[cur], 32, moff);
    const v16bf a1 = load_a_frag(sa[cur], 32, moff + 16);
    const v16bf b0 = load_bt_frag(sb[cur], 32, noff);
    const v16bf b1 = load_bt_frag(sb[cur], 32, noff + 16);
    acc[0] = wmma_bf16(a0, b0, acc[0]);
    acc[1] = wmma_bf16(a0, b1, acc[1]);
    acc[2] = wmma_bf16(a1, b0, acc[2]);
    acc[3] = wmma_bf16(a1, b1, acc[3]);
    __syncthreads();                          // done reading before next overwrite
  }
#pragma unroll
  for (int am = 0; am < 2; ++am)
#pragma unroll
    for (int bn = 0; bn < 2; ++bn)
#pragma unroll
      for (int r = 0; r < 8; ++r) {
        const int bi = mblk * 64 + moff + am * 16 + 8 * lh + r;
        const int o = nblk * 64 + noff + bn * 16 + lr;
        float v = acc[am * 2 + bn][r];
        v = (o < 500) ? fmaxf(v + bia[s * NB_ + 70 + o], 0.0f) : 0.0f; // zero K-pad for FC2
        h3[(size_t)(s * B_ + bi) * 512 + o] = f2bf(v);
      }
}

// ---------------- Stage 5: FC2 500->10 -> f32 logits ------------------------
__global__ __launch_bounds__(128) void k_aff2(const unsigned short* __restrict__ h3,
                                              const unsigned short* __restrict__ w4p,
                                              const float* __restrict__ bia,
                                              float* __restrict__ logits) {
  const int mblk = blockIdx.x, s = blockIdx.y;
  __shared__ alignas(16) unsigned short sa[64 * 32];
  __shared__ alignas(16) unsigned short sb[16 * 32];
  const int t = threadIdx.x;
  const unsigned short* h3s = h3 + (size_t)(s * B_ + mblk * 64) * 512;
  const unsigned short* w4s = w4p + (size_t)s * 8192;
  const int wv = t >> 5, lane = t & 31, lh = lane >> 4, lr = lane & 15;
  v8f acc = zero8();
  for (int kb = 0; kb < 512; kb += 32) {
    __syncthreads();
#pragma unroll
    for (int q = 0; q < 2; ++q) {
      const int v = t + q * 128, row = v >> 2, c8 = (v & 3) * 8;
      cp16(h3s + (size_t)row * 512 + kb + c8, sa + row * 32 + c8);
    }
    if (t < 64) {
      const int row = t >> 2, c8 = (t & 3) * 8;
      cp16(w4s + row * 512 + kb + c8, sb + row * 32 + c8);
    }
    wait_async<0>();
    __syncthreads();
    acc = wmma_bf16(load_a_frag(sa, 32, wv * 16), load_bt_frag(sb, 32, 0), acc);
  }
#pragma unroll
  for (int r = 0; r < 8; ++r) {
    if (lr < 10) {
      const int bi = mblk * 64 + wv * 16 + 8 * lh + r;
      logits[((size_t)bi * S_ + s) * 16 + lr] = acc[r] + bia[s * NB_ + 570 + lr];
    }
  }
}

// ---------------- Stage 6: log_softmax + mean over samples (deterministic) --
__global__ __launch_bounds__(128) void k_lsm(const float* __restrict__ logits,
                                             float* __restrict__ out) {
  __shared__ float slp[100 * 10];
  const int b = blockIdx.x, t = threadIdx.x;
  if (t < 100) {
    const float* lg = logits + ((size_t)b * S_ + t) * 16;
    float v[10], mx = -1e30f;
#pragma unroll
    for (int o = 0; o < 10; ++o) { v[o] = lg[o]; mx = fmaxf(mx, v[o]); }
    float sum = 0.0f;
#pragma unroll
    for (int o = 0; o < 10; ++o) sum += __expf(v[o] - mx);
    const float lse = mx + __logf(sum);
#pragma unroll
    for (int o = 0; o < 10; ++o) slp[t * 10 + o] = v[o] - lse;
  }
  __syncthreads();
  if (t < 10) {
    float a = 0.0f;
    for (int ss = 0; ss < 100; ++ss) a += slp[ss * 10 + t];  // fixed-order sum
    out[b * 10 + t] = a * 0.01f;
  }
}

extern "C" void kernel_launch(void* const* d_in, const int* in_sizes, int n_in,
                              void* d_out, int out_size, void* d_ws, size_t ws_size,
                              hipStream_t stream) {
  const float* x     = (const float*)d_in[0];
  const float* e     = (const float*)d_in[1];
  const float* mu_w  = (const float*)d_in[2];
  const float* rho_w = (const float*)d_in[3];
  const float* mu_b  = (const float*)d_in[4];
  const float* rho_b = (const float*)d_in[5];
  char* ws = (char*)d_ws;
  // Workspace layout (bytes, 16B aligned): ~200.4 MB total.
  unsigned short* w1p = (unsigned short*)(ws);                   //    204,800
  unsigned short* w2p = (unsigned short*)(ws + 204800);          //  6,553,600
  unsigned short* w3p = (unsigned short*)(ws + 6758400);         // 81,920,000
  unsigned short* w4p = (unsigned short*)(ws + 88678400);        //  1,638,400
  float*  bia    = (float*)(ws + 90316800);                      //    232,000
  unsigned short* h1 = (unsigned short*)(ws + 90548800);         // 73,728,000
  unsigned short* h2 = (unsigned short*)(ws + 164276800);        // 20,480,000
  unsigned short* h3 = (unsigned short*)(ws + 184756800);        // 13,107,200
  float* logits  = (float*)(ws + 197864000);                     //    819,200
  float* sp_w    = (float*)(ws + 198683200);                     //  1,722,000
  float* sp_b    = (float*)(ws + 200405200);                     //      2,320
  float* out = (float*)d_out;

  // Zero the padded weight region (graph-capture legal); pads stay zero.
  (void)hipMemsetAsync(ws, 0, 90316800, stream);
  k_softplus <<<dim3((NW_ + 255) / 256), 256, 0, stream>>>(rho_w, rho_b, sp_w, sp_b);
  k_transform<<<dim3(512, S_),          256, 0, stream>>>(e, mu_w, sp_w, mu_b, sp_b,
                                                          w1p, w2p, w3p, w4p, bia);
  k_conv1    <<<dim3(B_, S_, 2),        128, 0, stream>>>(x, w1p, bia, h1);
  k_conv2    <<<dim3(B_, S_),           128, 0, stream>>>(h1, w2p, bia, h2);
  k_aff1     <<<dim3(2, 8, S_),         128, 0, stream>>>(h2, w3p, bia, h3);
  k_aff2     <<<dim3(2, S_),            128, 0, stream>>>(h3, w4p, bia, logits);
  k_lsm      <<<dim3(B_),               128, 0, stream>>>(logits, out);
}
